// WordBiLSTMCRF_43525198578045
// MI455X (gfx1250) — matrix operand order repaired and checked by reading, not compiled, not run on hardware
//
#include <hip/hip_runtime.h>
#include <hip/hip_bf16.h>

// ---------------------------------------------------------------------------
// BiLSTM-CRF for MI455X (gfx1250, wave32, WMMA bf16 16x16x32, async-to-LDS)
//
//  1. prep_w_swz    : transpose+convert weights to bf16, swizzled fragment-
//                     major B-panel layout ([tile32x16][lane][16 bf16]) so a
//                     B fragment is one 32B contiguous per-lane load
//  2. add_bias      : bias = b_ih + b_hh (f32)
//  3. embed_gather  : xe_bf (row-major, K padded 300->320) + global prefetch
//  4. proj_gemm x2  : xw = xe @ W_ih^T + bias, WMMA bf16, tiles staged with
//                     global_load_async_to_lds_b128, double buffered
//  5. lstm x2       : persistent WGs; Whh^T resident in LDS (128KB); xw slice
//                     async-prefetched one timestep ahead (double buffer)
//  6. fc_gemm       : logits = [h_f|h_b] @ fc_W^T + fc_b
//  7. crf_kernel    : one wave per batch row, shuffle-based logsumexp
//  8. finalize      : mean -> d_out[0]
// ---------------------------------------------------------------------------

typedef __bf16 bf16_t;
typedef __attribute__((ext_vector_type(16))) __bf16 v16bf;
typedef __attribute__((ext_vector_type(8)))  float  v8f;

#define DEVINL __device__ __forceinline__

constexpr int B  = 64;
constexpr int T  = 512;
constexpr int E  = 300;
constexpr int H  = 128;
constexpr int C  = 20;
constexpr int BT = B * T;        // 32768
constexpr int G  = 4 * H;        // 512
constexpr int EP = 320;          // E padded
constexpr int CP = 32;           // C padded
constexpr int K2 = 2 * H;        // 256

#define WMMA_BF16(a, b, c) \
  __builtin_amdgcn_wmma_f32_16x16x32_bf16(false, (a), false, (b), (short)0, (c), false, false)

// ---- CDNA5 async-copy helpers (CDNA5 ISA, GVS mode) ----------
DEVINL unsigned lds_off_of(const void* p) {
  return (unsigned)(unsigned long long)p;  // low 32 bits = LDS offset
}
DEVINL void async_load_b128(unsigned lds_off, unsigned gl_off, const void* gbase) {
  asm volatile("global_load_async_to_lds_b128 %0, %1, %2"
               :: "v"(lds_off), "v"(gl_off),
                  "s"((unsigned long long)gbase)
               : "memory");
}
DEVINL void wait_async0() { asm volatile("s_wait_asynccnt 0x0" ::: "memory"); }

// ---- WMMA fragment helpers (A/B in contiguous k order; C/D per ISA layout) ------------------------
// A (16x32 bf16 MxK): lane l row M=l&15; e -> K = (l>>4)*16 + e
// -> one contiguous 16-element run per lane
DEVINL v16bf load_a_frag(const bf16_t* p, int ld) {
  int lane = threadIdx.x & 31;
  int row  = lane & 15;
  int koff = (lane >> 4) * 16;
  v16bf a;
#pragma unroll
  for (int e = 0; e < 16; ++e) {
    int k = koff + e;
    a[e] = p[row * ld + k];
  }
  return a;
}
// B panel pre-swizzled fragment-major: tile t holds 32 lanes x 16 bf16,
// element e of lane l = B[K = kt*32 + (l>>4)*16 + e][N = nt*16 + (l&15)].
// A fragment load is one 32B per-lane dereference (2x b128).
DEVINL v16bf load_b_swz(const bf16_t* panel, int tile) {
  int lane = threadIdx.x & 31;
  return *(const v16bf*)(panel + ((tile << 5) + lane) * 16);
}
// C/D (16x16 f32): N = lane&15, M = vgpr + 8*(lane>=16)
DEVINL void store_c_frag(float* p, int ld, v8f c) {
  int lane  = threadIdx.x & 31;
  int col   = lane & 15;
  int rbase = (lane >> 4) * 8;
#pragma unroll
  for (int r = 0; r < 8; ++r) p[(rbase + r) * ld + col] = c[r];
}

DEVINL float sigmoidf_(float x) { return 1.f / (1.f + __expf(-x)); }

// ---------------------------------------------------------------------------
// 1. Weight prep: src is (srcN x srcK) row-major f32; dst is swizzled bf16
//    B panel of logical shape (K x N) = transpose(src), zero padded.
__global__ void prep_w_swz(const float* __restrict__ src, bf16_t* __restrict__ dst,
                           int srcN, int srcK, int K, int N) {
  int idx = blockIdx.x * blockDim.x + threadIdx.x;
  if (idx >= K * N) return;
  int k = idx / N, n = idx % N;
  float v = (k < srcK && n < srcN) ? src[n * srcK + k] : 0.f;
  int tile = (k >> 5) * (N >> 4) + (n >> 4);
  int lane = (((k >> 4) & 1) << 4) | (n & 15);
  int e    = k & 15;
  dst[((tile << 5) + lane) * 16 + e] = (bf16_t)v;
}

__global__ void add_bias(const float* __restrict__ a, const float* __restrict__ b,
                         float* __restrict__ out, int n) {
  int i = blockIdx.x * blockDim.x + threadIdx.x;
  if (i < n) out[i] = a[i] + b[i];
}

// ---------------------------------------------------------------------------
// 3. Embedding gather -> bf16, K padded to EP. One block per token row.
__global__ void embed_gather(const int* __restrict__ x, const float* __restrict__ emb,
                             bf16_t* __restrict__ xe) {
  int row = blockIdx.x;
  int tok = x[row];
  if (row + 1 < BT) __builtin_prefetch(emb + (long)x[row + 1] * E, 0, 1);
  const float* er = emb + (long)tok * E;
  for (int e = threadIdx.x; e < EP; e += blockDim.x) {
    float v = (e < E) ? er[e] : 0.f;
    xe[(long)row * EP + e] = (bf16_t)v;
  }
}

// ---------------------------------------------------------------------------
// 4. Input projection, M=BT K=EP N=G. Block tile 128x64, 8 waves of 32x32.
//    Tiles staged with async-to-LDS b128, double buffered.
__global__ void proj_gemm(const bf16_t* __restrict__ A, const bf16_t* __restrict__ Bsw,
                          const float* __restrict__ bias, float* __restrict__ out,
                          int reverse) {
  __shared__ alignas(64) bf16_t As[2][128 * 32];  // 2 x 8KB
  __shared__ alignas(64) bf16_t Bs[2][4 * 512];   // 2 x 4KB (4 swizzled tiles)
  int tid  = threadIdx.x;
  int wave = tid >> 5;
  int lane = tid & 31;
  int m0   = blockIdx.x * 128;
  int n0   = blockIdx.y * 64;
  int wm   = (wave >> 1) * 32;
  int wn   = (wave & 1) * 32;

  auto stage = [&](int buf, int k0) {
    // A tile: 128 rows x 32 cols bf16 = 512 chunks of 16B
    for (int c = tid; c < 512; c += 256) {
      int row = c >> 2, co = (c & 3) * 8;
      unsigned goff = (unsigned)(((m0 + row) * EP + k0 + co) * 2);
      async_load_b128(lds_off_of(&As[buf][row * 32 + co]), goff, A);
    }
    // B: 4 contiguous swizzled tiles (4KB) starting at tile (k0/32)*32 + n0/16
    unsigned bbase = (unsigned)((((k0 >> 5) * (G >> 4) + (n0 >> 4)) << 9) * 2);
    {
      int c = tid;
      if (c < 256)
        async_load_b128(lds_off_of(&Bs[buf][c * 8]), bbase + c * 16, Bsw);
    }
  };

  stage(0, 0);
  wait_async0();
  __syncthreads();

  v8f acc[2][2] = {};
  int cur = 0;
  for (int k0 = 0; k0 < EP; k0 += 32) {
    if (k0 + 32 < EP) stage(cur ^ 1, k0 + 32);
#pragma unroll
    for (int fi = 0; fi < 2; ++fi) {
      v16bf a = load_a_frag(&As[cur][(wm + fi * 16) * 32], 32);
#pragma unroll
      for (int fj = 0; fj < 2; ++fj) {
        v16bf b = load_b_swz(Bs[cur], (wn >> 4) + fj);
        acc[fi][fj] = WMMA_BF16(a, b, acc[fi][fj]);
      }
    }
    wait_async0();
    __syncthreads();
    cur ^= 1;
  }

  int col = lane & 15, rbase = (lane >> 4) * 8;
#pragma unroll
  for (int fi = 0; fi < 2; ++fi)
#pragma unroll
    for (int fj = 0; fj < 2; ++fj)
#pragma unroll
      for (int r = 0; r < 8; ++r) {
        int gm = m0 + wm + fi * 16 + rbase + r;
        int gn = n0 + wn + fj * 16 + col;
        int bb = gm / T, tt = gm % T;
        int orow = reverse ? (bb * T + (T - 1 - tt)) : gm;
        out[(long)orow * G + gn] = acc[fi][fj][r] + bias[gn];
      }
}

// ---------------------------------------------------------------------------
// 5. LSTM recurrence. grid=4 (16 batch rows each), block=256 (8 waves).
//    LDS: Whh^T swizzled 128KB | xw double buffer 64KB | g 32KB | c 8KB | h 4KB
//    = 236KB of the 320KB WGP LDS. xw slice for step t+1 async-prefetched
//    while the WMMA GEMM for step t runs.
__global__ void lstm_kernel(const bf16_t* __restrict__ Wg, const float* __restrict__ xw,
                            bf16_t* __restrict__ hout, int reverse) {
  extern __shared__ char smem[];
  bf16_t* Ws   = (bf16_t*)smem;                    // 128*512 bf16, swizzled
  float*  xbuf = (float*)(smem + 131072);          // 2 x 16 x 512 f32
  float*  gbuf = (float*)(smem + 131072 + 65536);  // 16 x 512 f32
  float*  cbuf = (float*)(smem + 131072 + 65536 + 32768);
  bf16_t* hbuf = (bf16_t*)(smem + 131072 + 65536 + 32768 + 8192);

  int tid  = threadIdx.x;
  int wave = tid >> 5;
  int b0   = blockIdx.x * 16;

  { // stage swizzled Whh^T verbatim (contiguous b128 copies)
    const uint4* src = (const uint4*)Wg;
    uint4* dst = (uint4*)Ws;
    for (int i = tid; i < (H * G * 2) / 16; i += 256) dst[i] = src[i];
  }
  for (int i = tid; i < 16 * H; i += 256) { cbuf[i] = 0.f; hbuf[i] = (bf16_t)0.f; }

  auto stage_x = [&](int buf, int t) {
    // 16 rows x 512 f32 = 2048 chunks of 16B
    for (int c = tid; c < 2048; c += 256) {
      int row = c >> 7, o = (c & 127);
      unsigned goff = (unsigned)(((long)(b0 + row) * T + t) * G * 4 + o * 16);
      async_load_b128(lds_off_of(xbuf + buf * 8192 + row * G + o * 4), goff, xw);
    }
  };
  stage_x(0, 0);
  wait_async0();
  __syncthreads();

  int cur = 0;
  for (int t = 0; t < T; ++t) {
    if (t + 1 < T) stage_x(cur ^ 1, t + 1);

    // ---- g(16x512) = h(16x128) @ Whh^T(128x512), WMMA bf16 ----
    v8f acc[4] = {};
#pragma unroll
    for (int k0 = 0; k0 < H; k0 += 32) {
      v16bf a = load_a_frag(hbuf + k0, H);
#pragma unroll
      for (int j = 0; j < 4; ++j) {
        v16bf b = load_b_swz(Ws, (k0 >> 5) * (G >> 4) + wave * 4 + j);
        acc[j] = WMMA_BF16(a, b, acc[j]);
      }
    }
#pragma unroll
    for (int j = 0; j < 4; ++j) store_c_frag(gbuf + wave * 64 + j * 16, G, acc[j]);
    __syncthreads();

    // ---- gates + state update (16x128, 8 per thread), xw from LDS ----
    const float* xcur = xbuf + cur * 8192;
    for (int idx = tid; idx < 16 * H; idx += 256) {
      int r = idx >> 7, n = idx & (H - 1);
      const float* xr = xcur + r * G;
      float iv = sigmoidf_(gbuf[r * G + n]         + xr[n]);
      float fv = sigmoidf_(gbuf[r * G + H + n]     + xr[H + n]);
      float gv = tanhf    (gbuf[r * G + 2 * H + n] + xr[2 * H + n]);
      float ov = sigmoidf_(gbuf[r * G + 3 * H + n] + xr[3 * H + n]);
      float cc = fv * cbuf[idx] + iv * gv;
      float hh = ov * tanhf(cc);
      cbuf[idx] = cc;
      hbuf[idx] = (bf16_t)hh;
      int tout = reverse ? (T - 1 - t) : t;
      hout[((long)(b0 + r) * T + tout) * H + n] = (bf16_t)hh;
    }
    wait_async0();
    __syncthreads();
    cur ^= 1;
  }
}

// ---------------------------------------------------------------------------
// 6. FC: logits = [h_f|h_b] @ fcw(256x32 swizzled) + fc_b. One wave = 16 rows.
__global__ void fc_gemm(const bf16_t* __restrict__ hf, const bf16_t* __restrict__ hb,
                        const bf16_t* __restrict__ fcw, const float* __restrict__ fcb,
                        float* __restrict__ logits) {
  int tid  = threadIdx.x;
  int wave = tid >> 5;
  int m0   = blockIdx.x * 128 + wave * 16;

  v8f acc[2] = {};
#pragma unroll
  for (int k0 = 0; k0 < K2; k0 += 32) {
    const bf16_t* hsrc = (k0 < H) ? (hf + (long)m0 * H + k0)
                                  : (hb + (long)m0 * H + (k0 - H));
    v16bf a = load_a_frag(hsrc, H);
#pragma unroll
    for (int j = 0; j < 2; ++j) {
      v16bf b = load_b_swz(fcw, (k0 >> 5) * (CP >> 4) + j);
      acc[j] = WMMA_BF16(a, b, acc[j]);
    }
  }
  int lane = tid & 31, col = lane & 15, rbase = (lane >> 4) * 8;
#pragma unroll
  for (int j = 0; j < 2; ++j) {
    int n = j * 16 + col;
    if (n < C)
#pragma unroll
      for (int r = 0; r < 8; ++r)
        logits[(long)(m0 + rbase + r) * C + n] = acc[j][r] + fcb[n];
  }
}

// ---------------------------------------------------------------------------
// 7. CRF NLL per batch row: one wave, lane j owns state j (C=20 <= 32).
__global__ void crf_kernel(const float* __restrict__ logits, const int* __restrict__ tags,
                           const unsigned char* __restrict__ mask,
                           const float* __restrict__ trans, float* __restrict__ nll) {
  int b = blockIdx.x;
  int j = threadIdx.x;  // 0..31
  __shared__ float trs[C * C];
  for (int i = j; i < C * C; i += 32) trs[i] = trans[i];
  __syncthreads();

  float tcol[C];
#pragma unroll
  for (int i = 0; i < C; ++i) tcol[i] = 0.f;
  if (j < C)
    for (int i = 0; i < C; ++i) tcol[i] = trs[i * C + j];

  const float* lg = logits + (long)b * T * C;

  float num = 0.f;
  for (int t = j; t < T; t += 32) {
    if (mask[b * T + t]) {
      num += lg[t * C + tags[b * T + t]];
      if (t >= 1 && mask[b * T + t - 1])
        num += trs[tags[b * T + t - 1] * C + tags[b * T + t]];
    }
  }
  for (int off = 16; off; off >>= 1) num += __shfl_down(num, off);
  num = __shfl(num, 0);

  float a_j = (j < C) ? lg[j] : 0.f;
  for (int t = 1; t < T; ++t) {
    float m = -3.0e38f, s = 0.f;
#pragma unroll
    for (int i = 0; i < C; ++i) {
      float sc = __shfl(a_j, i) + tcol[i];
      float nm = fmaxf(m, sc);
      s = s * __expf(m - nm) + __expf(sc - nm);
      m = nm;
    }
    float la = m + __logf(s) + ((j < C) ? lg[t * C + j] : 0.f);
    if (j < C && mask[b * T + t]) a_j = la;
  }

  float m = -3.0e38f, s = 0.f;
#pragma unroll
  for (int i = 0; i < C; ++i) {
    float ai = __shfl(a_j, i);
    float nm = fmaxf(m, ai);
    s = s * __expf(m - nm) + __expf(ai - nm);
    m = nm;
  }
  if (j == 0) nll[b] = (m + __logf(s)) - num;
}

__global__ void finalize(const float* __restrict__ nll, float* __restrict__ out) {
  if (threadIdx.x == 0) {
    float s = 0.f;
    for (int i = 0; i < B; ++i) s += nll[i];
    out[0] = s / (float)B;
  }
}

// ---------------------------------------------------------------------------
extern "C" void kernel_launch(void* const* d_in, const int* in_sizes, int n_in,
                              void* d_out, int out_size, void* d_ws, size_t ws_size,
                              hipStream_t stream) {
  (void)in_sizes; (void)n_in; (void)out_size; (void)ws_size;

  const int*           x      = (const int*)d_in[0];
  const unsigned char* mask   = (const unsigned char*)d_in[1];
  const int*           tags   = (const int*)d_in[2];
  const float*         emb    = (const float*)d_in[3];
  const float*         W_ih_f = (const float*)d_in[4];
  const float*         W_hh_f = (const float*)d_in[5];
  const float*         b_ih_f = (const float*)d_in[6];
  const float*         b_hh_f = (const float*)d_in[7];
  const float*         W_ih_b = (const float*)d_in[8];
  const float*         W_hh_b = (const float*)d_in[9];
  const float*         b_ih_b = (const float*)d_in[10];
  const float*         b_hh_b = (const float*)d_in[11];
  const float*         fc_W   = (const float*)d_in[12];
  const float*         fc_b   = (const float*)d_in[13];
  const float*         transi = (const float*)d_in[14];

  float* out    = (float*)d_out;   // out[0]=loss, out[1..]=logits (BT*C)
  float* logits = out + 1;

  char* ws = (char*)d_ws;
  size_t off = 0;
  auto carve = [&](size_t bytes) {
    char* p = ws + off;
    off = (off + bytes + 255) & ~(size_t)255;
    return p;
  };
  bf16_t* xe_bf  = (bf16_t*)carve((size_t)BT * EP * 2);
  bf16_t* wihf_s = (bf16_t*)carve((size_t)EP * G * 2);
  bf16_t* wihb_s = (bf16_t*)carve((size_t)EP * G * 2);
  bf16_t* whhf_s = (bf16_t*)carve((size_t)H * G * 2);
  bf16_t* whhb_s = (bf16_t*)carve((size_t)H * G * 2);
  bf16_t* fcw_s  = (bf16_t*)carve((size_t)K2 * CP * 2);
  float*  bias_f = (float*)carve((size_t)G * 4);
  float*  bias_b = (float*)carve((size_t)G * 4);
  float*  xw_f   = (float*)carve((size_t)BT * G * 4);
  float*  xw_b   = (float*)carve((size_t)BT * G * 4);
  bf16_t* hf_bf  = (bf16_t*)carve((size_t)BT * H * 2);
  bf16_t* hb_bf  = (bf16_t*)carve((size_t)BT * H * 2);
  float*  nllbuf = (float*)carve((size_t)B * 4);

  {
    int n1 = EP * G;
    prep_w_swz<<<(n1 + 255) / 256, 256, 0, stream>>>(W_ih_f, wihf_s, G, E, EP, G);
    prep_w_swz<<<(n1 + 255) / 256, 256, 0, stream>>>(W_ih_b, wihb_s, G, E, EP, G);
    int n2 = H * G;
    prep_w_swz<<<(n2 + 255) / 256, 256, 0, stream>>>(W_hh_f, whhf_s, G, H, H, G);
    prep_w_swz<<<(n2 + 255) / 256, 256, 0, stream>>>(W_hh_b, whhb_s, G, H, H, G);
    int n3 = K2 * CP;
    prep_w_swz<<<(n3 + 255) / 256, 256, 0, stream>>>(fc_W, fcw_s, C, K2, K2, CP);
    add_bias<<<(G + 255) / 256, 256, 0, stream>>>(b_ih_f, b_hh_f, bias_f, G);
    add_bias<<<(G + 255) / 256, 256, 0, stream>>>(b_ih_b, b_hh_b, bias_b, G);
  }

  embed_gather<<<BT, 128, 0, stream>>>(x, emb, xe_bf);

  {
    dim3 grid(BT / 128, G / 64);
    proj_gemm<<<grid, 256, 0, stream>>>(xe_bf, wihf_s, bias_f, xw_f, 0);
    proj_gemm<<<grid, 256, 0, stream>>>(xe_bf, wihb_s, bias_b, xw_b, 1);
  }

  {
    size_t shmem = 131072 + 65536 + 32768 + 8192 + 4096;  // 241664 B
    lstm_kernel<<<B / 16, 256, shmem, stream>>>(whhf_s, xw_f, hf_bf, 0);
    lstm_kernel<<<B / 16, 256, shmem, stream>>>(whhb_s, xw_b, hb_bf, 1);
  }

  fc_gemm<<<BT / 128, 256, 0, stream>>>(hf_bf, hb_bf, fcw_s, fc_b, logits);

  crf_kernel<<<B, 32, 0, stream>>>(logits, tags, mask, transi, nllbuf);
  finalize<<<1, 32, 0, stream>>>(nllbuf, out);
}
